// LSTMPPOPolicy_14130442404261
// MI455X (gfx1250) — matrix operand order repaired and checked by reading, not compiled, not run on hardware
//
#include <hip/hip_runtime.h>
#include <hip/hip_bf16.h>
#include <cstdint>
#include <cstddef>

// ---------------------------------------------------------------------------
// CDNA5 (gfx1250) LSTM-PPO policy forward.
//  - all dense math: v_wmma_f32_16x16x32_bf16 (bf16 in, f32 accumulate)
//  - weights pre-converted once to bf16 [N,K] (K-contiguous) in workspace
//  - B tiles moved via gfx1250 async copy: global_load_async_to_lds_b128,
//    double-buffered, synchronized with s_wait_asynccnt
// ---------------------------------------------------------------------------

typedef __attribute__((ext_vector_type(16))) __bf16 v16bf;
typedef __attribute__((ext_vector_type(8)))  float  v8f;

#define BM 128
#define BN 64
#define KC 32

__device__ __forceinline__ float silu_f(float x) { return x / (1.f + __expf(-x)); }
__device__ __forceinline__ float sigm_f(float x) { return 1.f / (1.f + __expf(-x)); }

union FragU { uint4 u[2]; v16bf v; };

// One b128 (16-byte) async global->LDS copy per lane (ASYNCcnt-tracked).
__device__ __forceinline__ void async_copy_b128(uint32_t lds_off, const __bf16* gptr)
{
    uint64_t ga = (uint64_t)(uintptr_t)gptr;
    asm volatile("global_load_async_to_lds_b128 %0, %1, off"
                 :: "v"(lds_off), "v"(ga) : "memory");
}
__device__ __forceinline__ void wait_async_le1() {
    asm volatile("s_wait_asynccnt 0x1" ::: "memory");
}
__device__ __forceinline__ void wait_async_le0() {
    asm volatile("s_wait_asynccnt 0x0" ::: "memory");
}

// C[M,N] = A[M,K](fp32) @ Bw[N,K](bf16)^T  (+bias[n]) (+add[m*addStride+n]) (EPI==1 -> SiLU)
template <int EPI>
__global__ __launch_bounds__(256) void wmma_gemm(
    const float* __restrict__ A, const __bf16* __restrict__ Bw,
    const float* __restrict__ bias, const float* __restrict__ add, long addStride,
    float* __restrict__ C, int M, int N, int K)
{
    __shared__ __attribute__((aligned(16))) __bf16 As[BM][KC];       // 8 KB
    __shared__ __attribute__((aligned(16))) __bf16 Bs[2][BN][KC];    // 2 x 4 KB

    const int tid  = threadIdx.x;
    const int lane = tid & 31;
    const int wave = tid >> 5;            // 8 waves: 4 along M x 2 along N
    const int wm   = (wave >> 1) * 32;
    const int wn   = (wave & 1) * 32;
    const int m0   = blockIdx.y * BM;
    const int n0   = blockIdx.x * BN;
    const int lh   = lane >> 4;           // K-half selector per ISA fragment layout

    v8f acc[2][2] = {};

    // A staging: 128 rows, 2 threads/row, 16 fp32 -> bf16 each
    const int ar = tid >> 1;
    const int ac = (tid & 1) * 16;
    // B staging: 64 rows, 4 threads/row, one async b128 (8 bf16) each
    const int brow = tid >> 2;
    const int bseg = (tid & 3) * 8;
    const __bf16* bsrc = Bw + (size_t)(n0 + brow) * K + bseg;
    const uint32_t blds0 = (uint32_t)(uintptr_t)&Bs[0][brow][bseg];
    const uint32_t blds1 = (uint32_t)(uintptr_t)&Bs[1][brow][bseg];

    // prologue: async-load B chunk 0 into buffer 0
    async_copy_b128(blds0, bsrc);

    int buf = 0;
    for (int k0 = 0; k0 < K; k0 += KC) {
        // ---- stage A tile (fp32 -> bf16, vectorized cvt + ds_store_b128) ----
        #pragma unroll
        for (int j = 0; j < 16; ++j)
            As[ar][ac + j] = (__bf16)A[(size_t)(m0 + ar) * K + (k0 + ac + j)];

        // ---- issue async copy of next B chunk into the other buffer ----
        const bool have_next = (k0 + KC) < K;
        if (have_next) {
            async_copy_b128(buf ? blds0 : blds1, bsrc + (k0 + KC));
            __builtin_prefetch(&A[(size_t)(m0 + ar) * K + (k0 + KC + ac)], 0, 0);
            wait_async_le1();    // current chunk's async copy complete (in-order)
        } else {
            wait_async_le0();
        }
        __syncthreads();

        // ---- build fragments from LDS (two b128 loads each) ----
        FragU a[2], b[2];
        #pragma unroll
        for (int i = 0; i < 2; ++i) {
            const uint4* p = (const uint4*)&As[wm + i * 16 + (lane & 15)][0];
            a[i].u[0] = p[lh];          // K 0..15 half
            a[i].u[1] = p[2 + lh];      // K 16..31 half
        }
        #pragma unroll
        for (int j = 0; j < 2; ++j) {
            const uint4* q = (const uint4*)&Bs[buf][wn + j * 16 + (lane & 15)][0];
            b[j].u[0] = q[lh];          // K 0..15 half
            b[j].u[1] = q[2 + lh];      // K 16..31 half
        }
        #pragma unroll
        for (int i = 0; i < 2; ++i)
            #pragma unroll
            for (int j = 0; j < 2; ++j)
                acc[i][j] = __builtin_amdgcn_wmma_f32_16x16x32_bf16(
                    false, a[i].v, false, b[j].v, (short)0, acc[i][j], false, false);
        __syncthreads();
        buf ^= 1;
    }

    // ---- epilogue ----
    #pragma unroll
    for (int i = 0; i < 2; ++i) {
        #pragma unroll
        for (int j = 0; j < 2; ++j) {
            #pragma unroll
            for (int r = 0; r < 8; ++r) {
                const int gm = m0 + wm + i * 16 + r + 8 * lh;
                const int gn = n0 + wn + j * 16 + (lane & 15);
                float v = acc[i][j][r];
                if (bias) v += bias[gn];
                if (add)  v += add[(size_t)gm * (size_t)addStride + gn];
                if (EPI == 1) v = silu_f(v);
                C[(size_t)gm * N + gn] = v;
            }
        }
    }
}

// ---- weight prep ----
// out[i] = bf16(in[i])
__global__ void cvt_bf16_kernel(const float* __restrict__ in, __bf16* __restrict__ out, int n)
{
    int i = blockIdx.x * 256 + threadIdx.x;
    if (i < n) out[i] = (__bf16)in[i];
}
// in[k*N+n] -> out[n*K+k]  ([K,N] fp32 -> [N,K] bf16)
__global__ void cvt_t_bf16_kernel(const float* __restrict__ in, __bf16* __restrict__ out,
                                  int K, int N)
{
    int i = blockIdx.x * 256 + threadIdx.x;
    if (i < K * N) {
        int n = i / K;
        int k = i - n * K;
        out[i] = (__bf16)in[(size_t)k * N + n];
    }
}

// bias_sum[j] = b_ih[j] + b_hh[j]
__global__ void bias_sum_kernel(const float* __restrict__ a, const float* __restrict__ b,
                                float* __restrict__ o, int n)
{
    int i = blockIdx.x * blockDim.x + threadIdx.x;
    if (i < n) o[i] = a[i] + b[i];
}

// gates[B,4H] -> update h,c in place; write h into hseq[b,t,:]
__global__ __launch_bounds__(256) void lstm_elem(
    const float* __restrict__ gates, float* __restrict__ h, float* __restrict__ c,
    float* __restrict__ hseq, int t, int T, int Hd)
{
    int idx = blockIdx.x * 256 + threadIdx.x;   // 0 .. 256*1024-1
    int b = idx >> 10;
    int j = idx & 1023;
    const float* g = gates + (size_t)b * 4096;
    float ig = sigm_f(g[j]);
    float fg = sigm_f(g[1024 + j]);
    float gg = tanhf(g[2048 + j]);
    float og = sigm_f(g[3072 + j]);
    float cn = fg * c[idx] + ig * gg;
    float hn = og * tanhf(cn);
    c[idx] = cn;
    h[idx] = hn;
    hseq[((size_t)b * T + t) * Hd + j] = hn;
}

// per-row layernorm over H=1024 (256 threads x 4 elems)
__global__ __launch_bounds__(256) void layernorm_kernel(
    const float* __restrict__ X, const float* __restrict__ g, const float* __restrict__ b,
    float* __restrict__ Y, int Hd)
{
    __shared__ float red[256];
    const size_t row = blockIdx.x;
    const float* x = X + row * Hd;
    float v[4], s = 0.f;
    #pragma unroll
    for (int i = 0; i < 4; ++i) { v[i] = x[threadIdx.x + 256 * i]; s += v[i]; }
    red[threadIdx.x] = s;
    __syncthreads();
    for (int off = 128; off > 0; off >>= 1) {
        if (threadIdx.x < off) red[threadIdx.x] += red[threadIdx.x + off];
        __syncthreads();
    }
    const float mu = red[0] / Hd;
    __syncthreads();
    float vs = 0.f;
    #pragma unroll
    for (int i = 0; i < 4; ++i) { float d = v[i] - mu; vs += d * d; }
    red[threadIdx.x] = vs;
    __syncthreads();
    for (int off = 128; off > 0; off >>= 1) {
        if (threadIdx.x < off) red[threadIdx.x] += red[threadIdx.x + off];
        __syncthreads();
    }
    const float rstd = rsqrtf(red[0] / Hd + 1e-5f);
    #pragma unroll
    for (int i = 0; i < 4; ++i) {
        int col = threadIdx.x + 256 * i;
        Y[row * Hd + col] = (v[i] - mu) * rstd * g[col] + b[col];
    }
}

// N==1 heads: values = x.Wc + bc ; rew = x.Wr + br  (one block per row)
__global__ __launch_bounds__(256) void vec_heads(
    const float* __restrict__ X, const float* __restrict__ Wc, const float* __restrict__ bc,
    const float* __restrict__ Wr, const float* __restrict__ br,
    float* __restrict__ vout, float* __restrict__ rout, int Hd)
{
    __shared__ float r1[256], r2[256];
    const size_t row = blockIdx.x;
    const float* x = X + row * Hd;
    float s1 = 0.f, s2 = 0.f;
    for (int i = threadIdx.x; i < Hd; i += 256) {
        float xv = x[i];
        s1 += xv * Wc[i];
        s2 += xv * Wr[i];
    }
    r1[threadIdx.x] = s1;
    r2[threadIdx.x] = s2;
    __syncthreads();
    for (int off = 128; off > 0; off >>= 1) {
        if (threadIdx.x < off) {
            r1[threadIdx.x] += r1[threadIdx.x + off];
            r2[threadIdx.x] += r2[threadIdx.x + off];
        }
        __syncthreads();
    }
    if (threadIdx.x == 0) {
        vout[row] = r1[0] + bc[0];
        rout[row] = r2[0] + br[0];
    }
}

extern "C" void kernel_launch(void* const* d_in, const int* in_sizes, int n_in,
                              void* d_out, int out_size, void* d_ws, size_t ws_size,
                              hipStream_t stream)
{
    (void)in_sizes; (void)n_in; (void)out_size; (void)ws_size;

    const float* obs  = (const float*)d_in[0];
    const float* h0   = (const float*)d_in[1];
    const float* c0   = (const float*)d_in[2];
    const float* W1   = (const float*)d_in[3];
    const float* b1   = (const float*)d_in[4];
    const float* W2   = (const float*)d_in[5];
    const float* b2   = (const float*)d_in[6];
    const float* w_ih = (const float*)d_in[7];
    const float* w_hh = (const float*)d_in[8];
    const float* b_ih = (const float*)d_in[9];
    const float* b_hh = (const float*)d_in[10];
    const float* ln_g = (const float*)d_in[11];
    const float* ln_b = (const float*)d_in[12];
    const float* Wa   = (const float*)d_in[13];
    const float* ba   = (const float*)d_in[14];
    const float* Wc   = (const float*)d_in[15];
    const float* bc   = (const float*)d_in[16];
    const float* Wobs = (const float*)d_in[17];
    const float* bobs = (const float*)d_in[18];
    const float* Wr   = (const float*)d_in[19];
    const float* br   = (const float*)d_in[20];

    constexpr int Bb = 256, T = 128, OBS = 512, E = 1024, Hd = 1024, G4 = 4096, Ah = 64;
    constexpr size_t R = (size_t)Bb * T;   // 32768 rows

    // ---- workspace layout ----
    float* ws    = (float*)d_ws;
    float* gx    = ws;                           // R*4096   (512 MB)
    float* tmp1  = ws;                           // aliases gx prefix (R*E), dead before gx written
    float* enc   = ws + R * (size_t)G4;          // R*E
    float* xln   = enc;                          // aliases enc (enc dead after gx GEMM)
    float* hseq  = enc + R * (size_t)E;          // R*H
    float* hbuf  = hseq + R * (size_t)Hd;        // 256*1024
    float* cbuf  = hbuf + (size_t)Bb * Hd;       // 256*1024
    float* gates = cbuf + (size_t)Bb * Hd;       // 256*4096
    float* bsum  = gates + (size_t)Bb * G4;      // 4096
    // bf16 weights, all stored [N,K] K-contiguous
    __bf16* W1t   = (__bf16*)(bsum + G4);        // [1024][512]
    __bf16* W2t   = W1t + (size_t)E * OBS;       // [1024][1024]
    __bf16* wihB  = W2t + (size_t)E * E;         // [4096][1024]
    __bf16* whhB  = wihB + (size_t)G4 * E;       // [4096][1024]
    __bf16* WaT   = whhB + (size_t)G4 * Hd;      // [64][1024]
    __bf16* WobsT = WaT + (size_t)Ah * Hd;       // [512][1024]

    // ---- output layout (floats, concatenated in return order) ----
    float* out      = (float*)d_out;
    float* o_logits = out;                                     // R*64
    float* o_values = o_logits + R * (size_t)Ah;               // R
    float* o_obs    = o_values + R;                            // R*512
    float* o_rew    = o_obs + R * (size_t)OBS;                 // R
    float* o_hT     = o_rew + R;                               // 256*1024
    float* o_cT     = o_hT + (size_t)Bb * Hd;                  // 256*1024

    // ---- prep: bias sum, state init, bf16 weight conversion ----
    bias_sum_kernel<<<G4 / 256, 256, 0, stream>>>(b_ih, b_hh, bsum, G4);
    hipMemcpyAsync(hbuf, h0, (size_t)Bb * Hd * sizeof(float), hipMemcpyDeviceToDevice, stream);
    hipMemcpyAsync(cbuf, c0, (size_t)Bb * Hd * sizeof(float), hipMemcpyDeviceToDevice, stream);

    cvt_t_bf16_kernel<<<(OBS * E) / 256, 256, 0, stream>>>(W1, W1t, OBS, E);
    cvt_t_bf16_kernel<<<(E * E) / 256, 256, 0, stream>>>(W2, W2t, E, E);
    cvt_bf16_kernel<<<(G4 * E) / 256, 256, 0, stream>>>(w_ih, wihB, G4 * E);
    cvt_bf16_kernel<<<(G4 * Hd) / 256, 256, 0, stream>>>(w_hh, whhB, G4 * Hd);
    cvt_t_bf16_kernel<<<(Hd * Ah) / 256, 256, 0, stream>>>(Wa, WaT, Hd, Ah);
    cvt_t_bf16_kernel<<<(Hd * OBS) / 256, 256, 0, stream>>>(Wobs, WobsT, Hd, OBS);

    // ---- encoder: silu(silu(obs@W1+b1)@W2+b2) ----
    wmma_gemm<1><<<dim3(E / BN, (unsigned)(R / BM)), 256, 0, stream>>>(
        obs, W1t, b1, nullptr, 0, tmp1, (int)R, E, OBS);
    wmma_gemm<1><<<dim3(E / BN, (unsigned)(R / BM)), 256, 0, stream>>>(
        tmp1, W2t, b2, nullptr, 0, enc, (int)R, E, E);

    // ---- gx = enc @ w_ih^T + (b_ih + b_hh) ----
    wmma_gemm<0><<<dim3(G4 / BN, (unsigned)(R / BM)), 256, 0, stream>>>(
        enc, wihB, bsum, nullptr, 0, gx, (int)R, G4, E);

    // ---- LSTM scan over T ----
    for (int t = 0; t < T; ++t) {
        wmma_gemm<0><<<dim3(G4 / BN, Bb / BM), 256, 0, stream>>>(
            hbuf, whhB, nullptr, gx + (size_t)t * G4, (long)T * G4, gates, Bb, G4, Hd);
        lstm_elem<<<(Bb * Hd) / 256, 256, 0, stream>>>(gates, hbuf, cbuf, hseq, t, T, Hd);
    }

    // ---- layernorm + heads ----
    layernorm_kernel<<<(unsigned)R, 256, 0, stream>>>(hseq, ln_g, ln_b, xln, Hd);

    wmma_gemm<0><<<dim3(Ah / BN, (unsigned)(R / BM)), 256, 0, stream>>>(
        xln, WaT, ba, nullptr, 0, o_logits, (int)R, Ah, Hd);
    wmma_gemm<0><<<dim3(OBS / BN, (unsigned)(R / BM)), 256, 0, stream>>>(
        xln, WobsT, bobs, nullptr, 0, o_obs, (int)R, OBS, Hd);
    vec_heads<<<(unsigned)R, 256, 0, stream>>>(xln, Wc, bc, Wr, br, o_values, o_rew, Hd);

    hipMemcpyAsync(o_hT, hbuf, (size_t)Bb * Hd * sizeof(float), hipMemcpyDeviceToDevice, stream);
    hipMemcpyAsync(o_cT, cbuf, (size_t)Bb * Hd * sizeof(float), hipMemcpyDeviceToDevice, stream);
}
